// HatNet_89172111000126
// MI455X (gfx1250) — matrix operand order repaired and checked by reading, not compiled, not run on hardware
//
#include <hip/hip_runtime.h>
#include <hip/hip_bf16.h>

// ---------------- problem constants ----------------
#define B_    128
#define N_    548
#define DEG_  32
#define INC_  128
#define DIM_  256
#define OUT_  10
#define NT_   (B_ * N_)          // 70144
#define E_    (NT_ * DEG_)       // 2244608
#define K1_   274
#define K2_   137
#define HCAT_ (21 * DIM_)        // 5376
#define H2_   1344
#define BN_EPS 1e-5f

typedef __attribute__((ext_vector_type(16))) _Float16 v16h;
typedef __attribute__((ext_vector_type(8)))  _Float16 v8h;
typedef __attribute__((ext_vector_type(8)))  float    v8f;

__device__ __forceinline__ int imin(int a, int b) { return a < b ? a : b; }

// ================= WMMA GEMM =================
// C[M,N] = op(A)[M,K] @ op(B)[K,N]  (+ R) (+ bias[n]) (relu)
// op(A): TA ? A[k*lda+m] : A[m*lda+k]
// op(B): TB ? B[n*ldb+k] : B[k*ldb+n]
// batched via blockIdx.z with element strides sA/sB/sR/sC.
#define TM 64
#define TN 64
#define TK 32
#define LDSP (TK + 8)   // 40 halves = 80 bytes per row: 16B aligned, conflict-free b128 reads

template <bool TA, bool TB>
__global__ void __launch_bounds__(512)
wmma_gemm_kernel(const float* __restrict__ A, const float* __restrict__ B,
                 const float* __restrict__ bias, const float* __restrict__ R,
                 float* __restrict__ C,
                 int M, int N, int K, int lda, int ldb, int ldc,
                 long long sA, long long sB, long long sR, long long sC,
                 int doBias, int doRelu)
{
    // Both tiles stored fragment-major: second index is K.
    __shared__ _Float16 As[TM][LDSP];    // As[m][k]
    __shared__ _Float16 BsT[TN][LDSP];   // BsT[n][k]

    const int bz = blockIdx.z;
    A += (long long)bz * sA;
    B += (long long)bz * sB;
    C += (long long)bz * sC;
    if (R) R += (long long)bz * sR;

    const int tileM = blockIdx.y * TM;
    const int tileN = blockIdx.x * TN;
    const int tid   = threadIdx.x;
    const int lane  = tid & 31;
    const int wave  = tid >> 5;           // 16 waves
    const int wm    = (wave >> 2) * 16;   // wave tile row in block tile
    const int wn    = (wave & 3) * 16;    // wave tile col in block tile
    const int row16 = lane & 15;
    const int hi    = lane >> 4;          // lane group (0: lanes 0-15, 1: 16-31)

    v8f acc = {};

    for (int k0 = 0; k0 < K; k0 += TK) {
        // ===== phase 1: issue ALL global loads branch-free (clamped addresses) =====
        float va[4], ma[4], vb[4], mb[4];
        long long idxA0 = 0, idxB0 = 0;
#pragma unroll
        for (int i = 0; i < 4; ++i) {
            int t = tid + i * 512;
            int r, kk;
            if (TA) { kk = t >> 6; r = t & 63; }   // A[k*lda+m]: lanes sweep m
            else    { r = t >> 5; kk = t & 31; }   // A[m*lda+k]: lanes sweep k
            int m = tileM + r, k = k0 + kk;
            long long idx = TA ? (long long)imin(k, K - 1) * lda + imin(m, M - 1)
                               : (long long)imin(m, M - 1) * lda + imin(k, K - 1);
            if (i == 0) idxA0 = idx;
            ma[i] = (m < M && k < K) ? 1.f : 0.f;  // constant select, no branch
            va[i] = A[idx];                        // unconditional load
        }
#pragma unroll
        for (int i = 0; i < 4; ++i) {
            int t = tid + i * 512;
            int nn, kk;
            if (TB) { nn = t >> 5; kk = t & 31; }  // B[n*ldb+k]: lanes sweep k
            else    { kk = t >> 6; nn = t & 63; }  // B[k*ldb+n]: lanes sweep n
            int n = tileN + nn, k = k0 + kk;
            long long idx = TB ? (long long)imin(n, N - 1) * ldb + imin(k, K - 1)
                               : (long long)imin(k, K - 1) * ldb + imin(n, N - 1);
            if (i == 0) idxB0 = idx;
            mb[i] = (n < N && k < K) ? 1.f : 0.f;
            vb[i] = B[idx];
        }
        // one speculative prefetch pair per thread for the next K tile
        if (k0 + TK < K) {
            __builtin_prefetch(&A[idxA0 + (TA ? (long long)TK * lda : (long long)TK)], 0, 1);
            __builtin_prefetch(&B[idxB0 + (TB ? (long long)TK : (long long)TK * ldb)], 0, 1);
        }
        // ===== phase 2: convert + stage into LDS (mask-multiply keeps loads hoisted) =====
#pragma unroll
        for (int i = 0; i < 4; ++i) {
            int t = tid + i * 512;
            int r, kk;
            if (TA) { kk = t >> 6; r = t & 63; }
            else    { r = t >> 5; kk = t & 31; }
            As[r][kk] = (_Float16)(va[i] * ma[i]);
        }
#pragma unroll
        for (int i = 0; i < 4; ++i) {
            int t = tid + i * 512;
            int nn, kk;
            if (TB) { nn = t >> 5; kk = t & 31; }
            else    { kk = t >> 6; nn = t & 63; }
            BsT[nn][kk] = (_Float16)(vb[i] * mb[i]);
        }
        __syncthreads();

        // ---- A fragment: lane row M = wm+row16; K-base 0 (lanes 0-15) / 8 (16-31).
        // halves 0..7 -> K=base..base+7 ; halves 8..15 -> K=base+16..base+23.
        // Each chunk is 16B-aligned in LDS -> 2x ds_load_b128.
        {
            const int ab = hi ? 8 : 0;
            v8h a0 = *(const v8h*)&As[wm + row16][ab];
            v8h a1 = *(const v8h*)&As[wm + row16][ab + 16];
            v16h af = __builtin_shufflevector(a0, a1, 0, 1, 2, 3, 4, 5, 6, 7,
                                              8, 9, 10, 11, 12, 13, 14, 15);
            // ---- B fragment: lane col N = wn+row16; same K split as A (base, base+16).
            const int kb = hi ? 8 : 0;
            v8h b0 = *(const v8h*)&BsT[wn + row16][kb];
            v8h b1 = *(const v8h*)&BsT[wn + row16][kb + 16];
            v16h bf = __builtin_shufflevector(b0, b1, 0, 1, 2, 3, 4, 5, 6, 7,
                                              8, 9, 10, 11, 12, 13, 14, 15);

            acc = __builtin_amdgcn_wmma_f32_16x16x32_f16(
                /*neg_a=*/false, af, /*neg_b=*/false, bf,
                /*c_mod=*/(short)0, acc, /*reuse_a=*/false, /*reuse_b=*/false);
        }
        __syncthreads();
    }

    // epilogue: C/D layout — lane col N=lane&15, VGPR r -> row r + (hi?8:0)
    const int n = tileN + wn + row16;
    if (n < N) {
        float bv = doBias ? bias[n] : 0.f;
#pragma unroll
        for (int r = 0; r < 8; ++r) {
            int m = tileM + wm + r + (hi ? 8 : 0);
            if (m < M) {
                float v = acc[r] + bv;
                if (R) v += R[(long long)m * ldc + n];
                if (doRelu) v = fmaxf(v, 0.f);
                C[(long long)m * ldc + n] = v;
            }
        }
    }
}

// ================= elementwise / reduction kernels =================
__global__ void zero_kernel(float* __restrict__ p, long long n) {
    for (long long i = blockIdx.x * (long long)blockDim.x + threadIdx.x; i < n;
         i += (long long)gridDim.x * blockDim.x) p[i] = 0.f;
}

__global__ void copy_kernel(float* __restrict__ dst, const float* __restrict__ src, long long n) {
    for (long long i = blockIdx.x * (long long)blockDim.x + threadIdx.x; i < n;
         i += (long long)gridDim.x * blockDim.x) dst[i] = src[i];
}

// out[dst[e],c] += x[src[e],c]  (one block per edge)
__global__ void scatter_add_kernel(float* __restrict__ out, const float* __restrict__ x,
                                   const int* __restrict__ src, const int* __restrict__ dst, int C) {
    int e = blockIdx.x;
    int s = src[e], d = dst[e];
    const float* xs = x + (long long)s * C;
    float* od = out + (long long)d * C;
    for (int c = threadIdx.x; c < C; c += blockDim.x) atomicAdd(&od[c], xs[c]);
}

__global__ void adj_build_kernel(float* __restrict__ adj, const int* __restrict__ src,
                                 const int* __restrict__ dst, int E, int Nn) {
    for (long long e = blockIdx.x * (long long)blockDim.x + threadIdx.x; e < E;
         e += (long long)gridDim.x * blockDim.x) {
        int sg = src[e], dg = dst[e];
        int b = sg / Nn;
        atomicAdd(&adj[((long long)b * Nn + (sg % Nn)) * Nn + (dg % Nn)], 1.0f);
    }
}

__global__ void bn_reduce_kernel(const float* __restrict__ h, long long Mrows, int C,
                                 float* __restrict__ sums, float* __restrict__ sumsq) {
    long long per = (Mrows + gridDim.x - 1) / gridDim.x;
    long long r0 = (long long)blockIdx.x * per;
    long long r1 = r0 + per; if (r1 > Mrows) r1 = Mrows;
    for (int c = threadIdx.x; c < C; c += blockDim.x) {
        float s = 0.f, s2 = 0.f;
        for (long long r = r0; r < r1; ++r) { float v = h[r * C + c]; s += v; s2 += v * v; }
        atomicAdd(&sums[c], s);
        atomicAdd(&sumsq[c], s2);
    }
}

__global__ void bn_finalize_kernel(float* __restrict__ sums, float* __restrict__ sumsq,
                                   int C, float invM) {
    for (int c = blockIdx.x * blockDim.x + threadIdx.x; c < C; c += gridDim.x * blockDim.x) {
        float mu = sums[c] * invM;
        float var = sumsq[c] * invM - mu * mu;
        sums[c]  = mu;
        sumsq[c] = rsqrtf(var + BN_EPS);
    }
}

__global__ void bn_apply_relu_kernel(float* __restrict__ h, long long total, int C,
                                     const float* __restrict__ mu, const float* __restrict__ rs,
                                     const float* __restrict__ g, const float* __restrict__ be) {
    for (long long i = blockIdx.x * (long long)blockDim.x + threadIdx.x; i < total;
         i += (long long)gridDim.x * blockDim.x) {
        int c = (int)(i % C);
        float v = (h[i] - mu[c]) * rs[c] * g[c] + be[c];
        h[i] = fmaxf(v, 0.f);
    }
}

// sparse readout: hcat[b, off..] = {sum, sum/N, max} over N_ rows
__global__ void readout_sparse_kernel(const float* __restrict__ h, float* __restrict__ hcat, int off) {
    int b = blockIdx.x, c = threadIdx.x;
    const float* p = h + (long long)b * N_ * DIM_ + c;
    float s = 0.f, mx = -3.402823466e38f;
    for (int r = 0; r < N_; ++r) { float v = p[(long long)r * DIM_]; s += v; mx = fmaxf(mx, v); }
    float* o = hcat + (long long)b * HCAT_ + off;
    o[c] = s; o[DIM_ + c] = s * (1.0f / N_); o[2 * DIM_ + c] = mx;
}

// dense readout: {mean, sum, max} over `rows`
__global__ void readout_dense_kernel(const float* __restrict__ h, float* __restrict__ hcat,
                                     int off, int rows) {
    int b = blockIdx.x, c = threadIdx.x;
    const float* p = h + (long long)b * rows * DIM_ + c;
    float s = 0.f, mx = -3.402823466e38f;
    for (int r = 0; r < rows; ++r) { float v = p[(long long)r * DIM_]; s += v; mx = fmaxf(mx, v); }
    float* o = hcat + (long long)b * HCAT_ + off;
    o[c] = s / rows; o[DIM_ + c] = s; o[2 * DIM_ + c] = mx;
}

__global__ void softmax_kernel(float* __restrict__ s, int Kc) {
    long long row = blockIdx.x;
    float* p = s + row * Kc;
    __shared__ float red[128];
    float mx = -3.402823466e38f;
    for (int k = threadIdx.x; k < Kc; k += blockDim.x) mx = fmaxf(mx, p[k]);
    red[threadIdx.x] = mx; __syncthreads();
    for (int st = 64; st > 0; st >>= 1) {
        if (threadIdx.x < st) red[threadIdx.x] = fmaxf(red[threadIdx.x], red[threadIdx.x + st]);
        __syncthreads();
    }
    mx = red[0]; __syncthreads();
    float sum = 0.f;
    for (int k = threadIdx.x; k < Kc; k += blockDim.x) { float e = expf(p[k] - mx); p[k] = e; sum += e; }
    red[threadIdx.x] = sum; __syncthreads();
    for (int st = 64; st > 0; st >>= 1) {
        if (threadIdx.x < st) red[threadIdx.x] += red[threadIdx.x + st];
        __syncthreads();
    }
    float inv = 1.f / red[0];
    for (int k = threadIdx.x; k < Kc; k += blockDim.x) p[k] *= inv;
}

__global__ void trace_kernel(const float* __restrict__ m, int Kc, float* __restrict__ num) {
    int b = blockIdx.x;
    const float* p = m + (long long)b * Kc * Kc;
    __shared__ float red[128];
    float a = 0.f;
    for (int k = threadIdx.x; k < Kc; k += 128) a += p[(long long)k * Kc + k];
    red[threadIdx.x] = a; __syncthreads();
    for (int st = 64; st > 0; st >>= 1) {
        if (threadIdx.x < st) red[threadIdx.x] += red[threadIdx.x + st];
        __syncthreads();
    }
    if (!threadIdx.x) num[b] = red[0];
}

// den[b] = sum_n deg(n) * sum_k s[b,n,k]^2
__global__ void mincut_den_kernel(const float* __restrict__ adj, const float* __restrict__ s,
                                  int Nn, int Kc, float* __restrict__ den) {
    int b = blockIdx.x;
    const float* ab = adj + (long long)b * Nn * Nn;
    const float* sb = s + (long long)b * Nn * Kc;
    __shared__ float red[256];
    float acc = 0.f;
    for (int n = threadIdx.x; n < Nn; n += blockDim.x) {
        float d = 0.f;
        const float* ar = ab + (long long)n * Nn;
        for (int j = 0; j < Nn; ++j) d += ar[j];
        float ss = 0.f;
        const float* sr = sb + (long long)n * Kc;
        for (int k = 0; k < Kc; ++k) { float v = sr[k]; ss += v * v; }
        acc += d * ss;
    }
    red[threadIdx.x] = acc; __syncthreads();
    for (int st = 128; st > 0; st >>= 1) {
        if (threadIdx.x < st) red[threadIdx.x] += red[threadIdx.x + st];
        __syncthreads();
    }
    if (!threadIdx.x) den[b] = red[0];
}

__global__ void mc_loss_kernel(const float* __restrict__ num, const float* __restrict__ den,
                               float* __restrict__ slot) {
    __shared__ float red[128];
    red[threadIdx.x] = -(num[threadIdx.x] / den[threadIdx.x]);
    __syncthreads();
    for (int st = 64; st > 0; st >>= 1) {
        if (threadIdx.x < st) red[threadIdx.x] += red[threadIdx.x + st];
        __syncthreads();
    }
    if (!threadIdx.x) *slot = red[0] / (float)B_;
}

// ortho loss contribution for one graph, accumulated into *out
__global__ void ortho_kernel(const float* __restrict__ ss, int Kc, float* __restrict__ out, float invB) {
    int b = blockIdx.x;
    const float* p = ss + (long long)b * Kc * Kc;
    __shared__ float red[256];
    long long total = (long long)Kc * Kc;
    float acc = 0.f;
    for (long long i = threadIdx.x; i < total; i += blockDim.x) { float v = p[i]; acc += v * v; }
    red[threadIdx.x] = acc; __syncthreads();
    for (int st = 128; st > 0; st >>= 1) {
        if (threadIdx.x < st) red[threadIdx.x] += red[threadIdx.x + st];
        __syncthreads();
    }
    float inv = 1.f / sqrtf(red[0]);
    float ik = rsqrtf((float)Kc);
    __syncthreads();
    acc = 0.f;
    for (long long i = threadIdx.x; i < total; i += blockDim.x) {
        float v = p[i] * inv;
        if ((i % (Kc + 1)) == 0) v -= ik;   // diagonal
        acc += v * v;
    }
    red[threadIdx.x] = acc; __syncthreads();
    for (int st = 128; st > 0; st >>= 1) {
        if (threadIdx.x < st) red[threadIdx.x] += red[threadIdx.x + st];
        __syncthreads();
    }
    if (!threadIdx.x) atomicAdd(out, sqrtf(red[0]) * invB);
}

__global__ void zero_diag_kernel(float* __restrict__ m, int Kc) {
    int b = blockIdx.x;
    float* p = m + (long long)b * Kc * Kc;
    for (int k = threadIdx.x; k < Kc; k += blockDim.x) p[(long long)k * Kc + k] = 0.f;
}

__global__ void deg_kernel(const float* __restrict__ m, int Kc, float* __restrict__ d) {
    int b = blockIdx.x;
    const float* p = m + (long long)b * Kc * Kc;
    for (int k = threadIdx.x; k < Kc; k += blockDim.x) {
        float s = 0.f;
        const float* row = p + (long long)k * Kc;
        for (int j = 0; j < Kc; ++j) s += row[j];
        d[b * Kc + k] = sqrtf(s) + 1e-15f;
    }
}

__global__ void adjnorm_kernel(float* __restrict__ m, int Kc, const float* __restrict__ d, long long total) {
    for (long long i = blockIdx.x * (long long)blockDim.x + threadIdx.x; i < total;
         i += (long long)gridDim.x * blockDim.x) {
        long long kk = (long long)Kc * Kc;
        int b = (int)(i / kk);
        long long rem = i - (long long)b * kk;
        int r = (int)(rem / Kc), c = (int)(rem % Kc);
        m[i] = m[i] / (d[b * Kc + r] * d[b * Kc + c]);
    }
}

__global__ void loss_kernel(const float* __restrict__ logits, const long long* __restrict__ y,
                            const float* __restrict__ scal, float* __restrict__ out, int C) {
    __shared__ float red[128];
    int b = threadIdx.x;
    const float* p = logits + b * C;
    float mx = -3.402823466e38f;
    for (int c = 0; c < C; ++c) mx = fmaxf(mx, p[c]);
    float se = 0.f;
    for (int c = 0; c < C; ++c) se += expf(p[c] - mx);
    float lse = logf(se) + mx;
    int t = (int)y[b];
    red[b] = -(p[t] - lse);
    __syncthreads();
    for (int st = 64; st > 0; st >>= 1) {
        if (threadIdx.x < st) red[threadIdx.x] += red[threadIdx.x + st];
        __syncthreads();
    }
    if (!threadIdx.x)
        out[B_ * C] = red[0] / (float)B_ + scal[0] + scal[1] + scal[2] + scal[3];
}

// ================= host orchestration =================
static const long long NTD   = (long long)NT_ * DIM_;
static const long long BNN   = (long long)B_ * N_ * N_;
static const long long BNK1  = (long long)B_ * N_ * K1_;
static const long long BK1K1 = (long long)B_ * K1_ * K1_;
static const long long BK2K2 = (long long)B_ * K2_ * K2_;

struct NN { const float *w1, *b1, *g, *be, *w2, *b2; };

static NN getnn(void* const* d_in, int base) {
    return { (const float*)d_in[base + 0], (const float*)d_in[base + 1],
             (const float*)d_in[base + 2], (const float*)d_in[base + 3],
             (const float*)d_in[base + 4], (const float*)d_in[base + 5] };
}

static void gemm(hipStream_t st,
                 const float* A, int lda, long long sA, int tA,
                 const float* B, int ldb, long long sB, int tB,
                 const float* bias, const float* R, long long sR,
                 float* C, int ldc, long long sC,
                 int M, int N, int K, int batch, int relu) {
    dim3 grid((N + TN - 1) / TN, (M + TM - 1) / TM, batch);
    int doBias = bias != nullptr;
    if (tA) {
        if (tB) wmma_gemm_kernel<true, true><<<grid, 512, 0, st>>>(
            A, B, bias, R, C, M, N, K, lda, ldb, ldc, sA, sB, sR, sC, doBias, relu);
        else    wmma_gemm_kernel<true, false><<<grid, 512, 0, st>>>(
            A, B, bias, R, C, M, N, K, lda, ldb, ldc, sA, sB, sR, sC, doBias, relu);
    } else {
        if (tB) wmma_gemm_kernel<false, true><<<grid, 512, 0, st>>>(
            A, B, bias, R, C, M, N, K, lda, ldb, ldc, sA, sB, sR, sC, doBias, relu);
        else    wmma_gemm_kernel<false, false><<<grid, 512, 0, st>>>(
            A, B, bias, R, C, M, N, K, lda, ldb, ldc, sA, sB, sR, sC, doBias, relu);
    }
}

static void run_bn_relu(hipStream_t st, float* h, long long Mrows, int C,
                        const float* g, const float* be, float* sums) {
    float* sumsq = sums + 2048;
    zero_kernel<<<16, 256, 0, st>>>(sums, 4096);
    bn_reduce_kernel<<<256, 256, 0, st>>>(h, Mrows, C, sums, sumsq);
    bn_finalize_kernel<<<8, 256, 0, st>>>(sums, sumsq, C, 1.0f / (float)Mrows);
    bn_apply_relu_kernel<<<2048, 256, 0, st>>>(h, Mrows * C, C, sums, sumsq, g, be);
}

static void gin_sparse(hipStream_t st, const float* xin, int Cin, const NN& p,
                       const int* src, const int* dst,
                       float* t0, float* t1, float* hout, float* sums) {
    copy_kernel<<<2048, 256, 0, st>>>(t0, xin, (long long)NT_ * Cin);
    scatter_add_kernel<<<E_, 256, 0, st>>>(t0, xin, src, dst, Cin);
    gemm(st, t0, Cin, 0, 0, p.w1, Cin, 0, 1, p.b1, nullptr, 0,
         t1, DIM_, 0, NT_, DIM_, Cin, 1, 0);
    run_bn_relu(st, t1, NT_, DIM_, p.g, p.be, sums);
    gemm(st, t1, DIM_, 0, 0, p.w2, DIM_, 0, 1, p.b2, nullptr, 0,
         hout, DIM_, 0, NT_, DIM_, DIM_, 1, 1);
}

static void gin_dense(hipStream_t st, const float* xin, int rows, const float* adjp, const NN& p,
                      float* t0, float* t1, float* hout, float* sums) {
    // t0 = adjp @ xin + xin
    gemm(st, adjp, rows, (long long)rows * rows, 0,
         xin, DIM_, (long long)rows * DIM_, 0,
         nullptr, xin, (long long)rows * DIM_,
         t0, DIM_, (long long)rows * DIM_, rows, DIM_, rows, B_, 0);
    long long Mr = (long long)B_ * rows;
    gemm(st, t0, DIM_, 0, 0, p.w1, DIM_, 0, 1, p.b1, nullptr, 0,
         t1, DIM_, 0, (int)Mr, DIM_, DIM_, 1, 0);
    run_bn_relu(st, t1, Mr, DIM_, p.g, p.be, sums);
    gemm(st, t1, DIM_, 0, 0, p.w2, DIM_, 0, 1, p.b2, nullptr, 0,
         hout, DIM_, 0, (int)Mr, DIM_, DIM_, 1, 1);
}

static void mincut_pool(hipStream_t st, const float* xin, int rows, int Kc,
                        const float* adjcur, const float* pw, const float* pbv,
                        float* sbuf, float* adjs, float* outx, float* outadj,
                        float* dvec, float* num, float* den, float* scal, int idx) {
    // s = xin @ pw^T + pb ; softmax rows
    gemm(st, xin, DIM_, (long long)rows * DIM_, 0, pw, DIM_, 0, 1, pbv, nullptr, 0,
         sbuf, Kc, (long long)rows * Kc, rows, Kc, DIM_, B_, 0);
    softmax_kernel<<<B_ * rows, 128, 0, st>>>(sbuf, Kc);
    // outx = s^T @ xin
    gemm(st, sbuf, Kc, (long long)rows * Kc, 1, xin, DIM_, (long long)rows * DIM_, 0,
         nullptr, nullptr, 0, outx, DIM_, (long long)Kc * DIM_, Kc, DIM_, rows, B_, 0);
    // adjs = adj @ s
    gemm(st, adjcur, rows, (long long)rows * rows, 0, sbuf, Kc, (long long)rows * Kc, 0,
         nullptr, nullptr, 0, adjs, Kc, (long long)rows * Kc, rows, Kc, rows, B_, 0);
    // outadj = s^T @ adjs
    gemm(st, sbuf, Kc, (long long)rows * Kc, 1, adjs, Kc, (long long)rows * Kc, 0,
         nullptr, nullptr, 0, outadj, Kc, (long long)Kc * Kc, Kc, Kc, rows, B_, 0);
    trace_kernel<<<B_, 128, 0, st>>>(outadj, Kc, num);
    mincut_den_kernel<<<B_, 256, 0, st>>>(adjcur, sbuf, rows, Kc, den);
    mc_loss_kernel<<<1, 128, 0, st>>>(num, den, scal + idx);
    // ss = s^T @ s (reuse adjs), ortho loss
    gemm(st, sbuf, Kc, (long long)rows * Kc, 1, sbuf, Kc, (long long)rows * Kc, 0,
         nullptr, nullptr, 0, adjs, Kc, (long long)Kc * Kc, Kc, Kc, rows, B_, 0);
    ortho_kernel<<<B_, 256, 0, st>>>(adjs, Kc, scal + 2 + idx, 1.0f / (float)B_);
    // zero diag + symmetric degree normalization
    zero_diag_kernel<<<B_, 256, 0, st>>>(outadj, Kc);
    deg_kernel<<<B_, 256, 0, st>>>(outadj, Kc, dvec);
    adjnorm_kernel<<<2048, 256, 0, st>>>(outadj, Kc, dvec, (long long)B_ * Kc * Kc);
}

extern "C" void kernel_launch(void* const* d_in, const int* in_sizes, int n_in,
                              void* d_out, int out_size, void* d_ws, size_t ws_size,
                              hipStream_t stream) {
    (void)in_sizes; (void)n_in; (void)out_size; (void)ws_size;
    const float* x = (const float*)d_in[0];
    const int* ei  = (const int*)d_in[1];
    const int* src = ei;
    const int* dst = ei + E_;
    const long long* y = (const long long*)d_in[3];
    NN nn1 = getnn(d_in, 4),  nn2 = getnn(d_in, 10), nn3 = getnn(d_in, 16);
    NN nn4 = getnn(d_in, 22), nn5 = getnn(d_in, 28);
    const float* p1w = (const float*)d_in[34]; const float* p1b = (const float*)d_in[35];
    const float* p2w = (const float*)d_in[36]; const float* p2b = (const float*)d_in[37];
    const float* hw1 = (const float*)d_in[38]; const float* hb1 = (const float*)d_in[39];
    const float* hg  = (const float*)d_in[40]; const float* hbe = (const float*)d_in[41];
    const float* hw2 = (const float*)d_in[42]; const float* hb2 = (const float*)d_in[43];

    // workspace arena
    float* W = (float*)d_ws;
    long long off = 0;
    auto alloc = [&](long long n) { float* p = W + off; off += n; return p; };
    float* bufA  = alloc(NTD);
    float* bufB  = alloc(NTD);
    float* h1    = alloc(NTD);
    float* h2    = alloc(NTD);
    float* adj   = alloc(BNN);
    float* sbuf  = alloc(BNK1);
    float* adjs  = alloc(BNK1);
    float* adjp1 = alloc(BK1K1);
    float* adjp2 = alloc(BK2K2);
    float* hcat  = alloc((long long)B_ * HCAT_);
    float* zbuf  = alloc((long long)B_ * H2_);
    float* sums  = alloc(4096);                // sums + sumsq
    float* dvec  = alloc((long long)B_ * K1_);
    float* num   = alloc(B_);
    float* den   = alloc(B_);
    float* scal  = alloc(16);

    hipStream_t st = stream;
    zero_kernel<<<1, 64, 0, st>>>(scal, 16);

    // ---- sparse GIN layers ----
    gin_sparse(st, x, INC_, nn1, src, dst, bufA, bufB, h1, sums);
    readout_sparse_kernel<<<B_, DIM_, 0, st>>>(h1, hcat, 0 * DIM_);
    gin_sparse(st, h1, DIM_, nn2, src, dst, bufA, bufB, h2, sums);
    readout_sparse_kernel<<<B_, DIM_, 0, st>>>(h2, hcat, 3 * DIM_);

    // ---- dense adjacency ----
    zero_kernel<<<8192, 256, 0, st>>>(adj, BNN);
    adj_build_kernel<<<4096, 256, 0, st>>>(adj, src, dst, E_, N_);

    // ---- pool 1 (548 -> 274), pooled x into h1 ----
    mincut_pool(st, h2, N_, K1_, adj, p1w, p1b, sbuf, adjs, h1, adjp1, dvec, num, den, scal, 0);
    readout_dense_kernel<<<B_, DIM_, 0, st>>>(h1, hcat, 6 * DIM_, K1_);

    // ---- dense GIN 3 (out -> bufA) ----
    gin_dense(st, h1, K1_, adjp1, nn3, bufA, bufB, bufA, sums);
    readout_dense_kernel<<<B_, DIM_, 0, st>>>(bufA, hcat, 9 * DIM_, K1_);

    // ---- dense GIN 4 (out -> bufB) ----
    gin_dense(st, bufA, K1_, adjp1, nn4, bufB, h2, bufB, sums);
    readout_dense_kernel<<<B_, DIM_, 0, st>>>(bufB, hcat, 12 * DIM_, K1_);

    // ---- pool 2 (274 -> 137), pooled x into h1 ----
    mincut_pool(st, bufB, K1_, K2_, adjp1, p2w, p2b, sbuf, adjs, h1, adjp2, dvec, num, den, scal, 1);
    readout_dense_kernel<<<B_, DIM_, 0, st>>>(h1, hcat, 15 * DIM_, K2_);

    // ---- dense GIN 5 (out -> bufA) ----
    gin_dense(st, h1, K2_, adjp2, nn5, bufA, bufB, bufA, sums);
    readout_dense_kernel<<<B_, DIM_, 0, st>>>(bufA, hcat, 18 * DIM_, K2_);

    // ---- head ----
    gemm(st, hcat, HCAT_, 0, 0, hw1, HCAT_, 0, 1, hb1, nullptr, 0,
         zbuf, H2_, 0, B_, H2_, HCAT_, 1, 0);
    run_bn_relu(st, zbuf, B_, H2_, hg, hbe, sums);
    float* logits = (float*)d_out;
    gemm(st, zbuf, H2_, 0, 0, hw2, H2_, 0, 1, hb2, nullptr, 0,
         logits, OUT_, 0, B_, OUT_, H2_, 1, 0);
    loss_kernel<<<1, 128, 0, st>>>(logits, y, scal, logits, OUT_);
}
